// DU_LM_SB_54022098649886
// MI455X (gfx1250) — matrix operand order repaired and checked by reading, not compiled, not run on hardware
//
#include <hip/hip_runtime.h>
#include <math.h>

// ---------------- Problem constants (from setup_inputs) ----------------
#define NR    256
#define NT    256
#define NHT   512          // 2*NT
#define NRB   1024         // 2*NT*rb
#define BSZ   4096
#define TSTEPS 50
#define AUGC  1025         // 512 G cols + 512 Ht cols + 1 ytilde col
#define AUGS  1032         // padded row stride (floats)
#define QAM_VAR 10.0f      // 2*(16-1)/3

// LDS staging geometry for the SB step
#define BSTRIDE 528        // bytes per staged x row (256 bf16 = 512B + 16B pad)
#define ASTRIDE 80         // bytes per staged J row (32 bf16 = 64B + 16B pad)
#define BBYTES  (32 * BSTRIDE)   // 16896
#define ABYTES  (64 * ASTRIDE)   // 5120
#define SMEMB   (BBYTES + ABYTES)

typedef __attribute__((ext_vector_type(16))) __bf16 v16bf;
typedef __attribute__((ext_vector_type(8)))  __bf16 v8bf;
typedef __attribute__((ext_vector_type(8)))  float  v8f;

__device__ __forceinline__ unsigned short f2bf(float f) {
  unsigned int u = __float_as_uint(f);
  u += 0x7FFFu + ((u >> 16) & 1u);          // round-to-nearest-even
  return (unsigned short)(u >> 16);
}
__device__ __forceinline__ float sigmoidf_(float v) { return 1.0f / (1.0f + __expf(-v)); }
__device__ __forceinline__ float swishf_(float v)   { return v * sigmoidf_(v); }
__device__ __forceinline__ float phi_s(float x)     { return swishf_(x + 1.0f) - swishf_(x - 1.0f) - 1.0f; }
__device__ __forceinline__ float psi_s(float x)     { return sigmoidf_(100.0f * (fabsf(x) - 1.01f)); }

// gfx1250 async global->LDS copy of 16 bytes per lane (ASYNCcnt-tracked)
__device__ __forceinline__ void async_ld16(unsigned lds_off, const void* gaddr) {
  asm volatile("global_load_async_to_lds_b128 %0, %1, off"
               :: "v"(lds_off), "v"(gaddr) : "memory");
}

// ---------------- Phase 1: LM setup ----------------

// Ht = [[Hr, -Hi],[Hi, Hr]]  (512x512), ytilde = [yr; yi]
__global__ void k_build_ht(const float* __restrict__ Hr, const float* __restrict__ Hi,
                           const float* __restrict__ yr, const float* __restrict__ yi,
                           float* __restrict__ Ht, float* __restrict__ ytl) {
  int idx = blockIdx.x * 256 + threadIdx.x;
  if (idx < NHT) ytl[idx] = (idx < NR) ? yr[idx] : yi[idx - NR];
  if (idx >= NHT * NHT) return;
  int i = idx >> 9, j = idx & 511;
  float v;
  if (i < NR) v = (j < NT) ? Hr[i * NT + j] : -Hi[i * NT + (j - NT)];
  else        v = (j < NT) ? Hi[(i - NR) * NT + j] : Hr[(i - NR) * NT + (j - NT)];
  Ht[idx] = v;
}

// Aug = [ G | Ht | ytilde | pad ],  G = Ht*Ht^T + lam*I
__global__ void k_build_aug(const float* __restrict__ Ht, const float* __restrict__ ytl,
                            const float* __restrict__ lamp, float* __restrict__ Aug) {
  int idx = blockIdx.x * 256 + threadIdx.x;
  if (idx >= NHT * AUGS) return;
  int r = idx / AUGS, c = idx - r * AUGS;
  float v;
  if (c < NHT) {
    const float* a = Ht + (size_t)r * NHT;
    const float* b = Ht + (size_t)c * NHT;
    float acc = 0.0f;
    for (int k = 0; k < NHT; ++k) acc = fmaf(a[k], b[k], acc);
    v = acc + ((r == c) ? lamp[0] : 0.0f);
  } else if (c < 2 * NHT) {
    v = Ht[(size_t)r * NHT + (c - NHT)];
  } else if (c == 2 * NHT) {
    v = ytl[r];
  } else {
    v = 0.0f;
  }
  Aug[idx] = v;
}

// Gauss-Jordan on [G | Ht | y]: single workgroup, SPD so no pivoting.
__global__ __launch_bounds__(1024) void k_gauss_jordan(float* __restrict__ Aug) {
  __shared__ float prow[AUGC];
  __shared__ float fcol[NHT];
  __shared__ float sinv;
  const int tid = threadIdx.x;
  for (int p = 0; p < NHT; ++p) {
    if (tid == 0) sinv = 1.0f / Aug[(size_t)p * AUGS + p];
    __syncthreads();
    const float inv = sinv;
    for (int c = tid; c < AUGC; c += 1024) prow[c] = Aug[(size_t)p * AUGS + c] * inv;
    if (tid < NHT) fcol[tid] = Aug[(size_t)tid * AUGS + p];
    __syncthreads();
    for (int c = tid; c < AUGC; c += 1024) Aug[(size_t)p * AUGS + c] = prow[c];
    const int r = tid >> 1, half = tid & 1;
    if (r != p) {
      const float f = fcol[r];
      const int c0 = half * 513;
      const int c1 = (c0 + 513 < AUGC) ? (c0 + 513) : AUGC;
      float* rowp = Aug + (size_t)r * AUGS;
      for (int c = c0; c < c1; ++c) rowp[c] = fmaf(-f, prow[c], rowp[c]);
    }
    __syncthreads();
  }
}

// Kmat = Ht^T * (G^{-1}Ht)   (512x512)
__global__ void k_kmat(const float* __restrict__ Ht, const float* __restrict__ Aug,
                       float* __restrict__ Kmat) {
  int idx = blockIdx.x * 256 + threadIdx.x;
  if (idx >= NHT * NHT) return;
  int m = idx >> 9, n = idx & 511;
  float acc = 0.0f;
  for (int r = 0; r < NHT; ++r)
    acc = fmaf(Ht[(size_t)r * NHT + m], Aug[(size_t)r * AUGS + NHT + n], acc);
  Kmat[idx] = acc;
}

__global__ void k_stats_init(float* __restrict__ stats) { stats[0] = 0.0f; }

// J[i,i2] = -(2/qam_var)/lam * w_j * w_k * Kmat[m,n], zero diag; accumulate ||J||_F^2
__global__ __launch_bounds__(256) void k_expand_J(const float* __restrict__ Kmat,
                                                  const float* __restrict__ lamp,
                                                  unsigned short* __restrict__ Jb,
                                                  float* __restrict__ stats) {
  __shared__ float red[256];
  int idx = blockIdx.x * 256 + threadIdx.x;           // 0 .. NRB*NRB-1
  int i = idx >> 10, i2 = idx & 1023;
  int m = i & 511, jb = i >> 9;
  int n = i2 & 511, kb = i2 >> 9;
  float wj = jb ? 1.0f : 2.0f;
  float wk = kb ? 1.0f : 2.0f;
  float val = (i == i2) ? 0.0f
            : (-(2.0f / QAM_VAR) / lamp[0]) * wj * wk * Kmat[(size_t)m * NHT + n];
  Jb[idx] = f2bf(val);
  red[threadIdx.x] = val * val;
  __syncthreads();
  for (int s = 128; s > 0; s >>= 1) {
    if (threadIdx.x < s) red[threadIdx.x] += red[threadIdx.x + s];
    __syncthreads();
  }
  if (threadIdx.x == 0) atomicAdd(stats, red[0]);
}

// c0 into stats[1]; h vector (1024)
__global__ __launch_bounds__(512) void k_c0_h(const float* __restrict__ Ht,
                                              const float* __restrict__ Aug,
                                              const float* __restrict__ lamp,
                                              float* __restrict__ stats,
                                              float* __restrict__ hvec) {
  int m = threadIdx.x;
  float acc = 0.0f;
  for (int r = 0; r < NHT; ++r)
    acc = fmaf(Ht[(size_t)r * NHT + m], Aug[(size_t)r * AUGS + 2 * NHT], acc);
  float s = 2.0f / (lamp[0] * sqrtf(QAM_VAR));
  hvec[m]       = 2.0f * acc * s;   // w = 2 block
  hvec[NHT + m] = 1.0f * acc * s;   // w = 1 block
  if (m == 0) stats[1] = 0.5f * sqrtf((float)(NRB - 1)) * rsqrtf(stats[0]);
}

// x = 0.02*(x0-0.5); yv = 0.02*(y0-0.5); xb = bf16(x)
__global__ void k_init_xy(const float* __restrict__ x0, const float* __restrict__ y0,
                          float* __restrict__ X, float* __restrict__ YV,
                          unsigned short* __restrict__ XB) {
  size_t idx = (size_t)blockIdx.x * 256 + threadIdx.x;
  if (idx >= (size_t)NRB * BSZ) return;
  float x = 0.02f * (x0[idx] - 0.5f);
  float y = 0.02f * (y0[idx] - 0.5f);
  X[idx] = x; YV[idx] = y; XB[idx] = f2bf(x);
}

// ---------------- Phase 2: fused SB step (async-LDS staged GEMM + elementwise) ----
// Block: 256 threads = 8 waves as 2(M) x 4(N). Wave tile: 32x64. Block tile: 64x256.
// K staged in 32-wide chunks, double-buffered in LDS via global_load_async_to_lds.
__global__ __launch_bounds__(256) void k_sb_step(
    const unsigned short* __restrict__ Jb_u,
    const unsigned short* __restrict__ xbsrc_u,
    unsigned short* __restrict__ xbdst_u,
    float* __restrict__ X, float* __restrict__ YV,
    const float* __restrict__ hvec, const float* __restrict__ stats,
    const float* __restrict__ etap, const float* __restrict__ Delta,
    int t, float ak) {
  const __bf16* Jb    = (const __bf16*)Jb_u;
  const __bf16* xbsrc = (const __bf16*)xbsrc_u;

  __shared__ __align__(16) unsigned char sm[2][SMEMB];

  const int tid  = threadIdx.x;
  const int lane = tid & 31;
  const int wave = tid >> 5;
  const int wm = wave >> 2;                 // 0..1
  const int wn = wave & 3;                  // 0..3
  const int MoBlk = blockIdx.y * 64;
  const int NoBlk = blockIdx.x * 256;
  const int Mo = MoBlk + wm * 32;
  const int No = NoBlk + wn * 64;

  const float dk    = Delta[t];
  const float ec0   = etap[0] * stats[1];
  const float om_ak = 1.0f - ak;

  // LDS base offsets (generic LDS pointers carry the LDS offset in bits [31:0])
  const unsigned sb[2] = { (unsigned)(size_t)&sm[0][0], (unsigned)(size_t)&sm[1][0] };

  // Stage one K-chunk (x: 32 rows x 256 cols; J: 64 rows x 32 cols) into buffer.
  // Exactly 5 async instructions per wave -> s_wait_asynccnt 5 isolates a buffer.
  auto stage = [&](int ko, int buf) {
    const unsigned base = sb[buf];
#pragma unroll
    for (int q = 0; q < 4; ++q) {
      const int row = q * 8 + wave;                     // 0..31 (one full row/wave)
      const void* g = (const void*)(xbsrc + (size_t)(ko + row) * BSZ + NoBlk + lane * 8);
      async_ld16(base + row * BSTRIDE + lane * 16, g);
    }
    {
      const int row = tid >> 2;                         // 0..63
      const int cu  = tid & 3;                          // 0..3 (16B units)
      const void* g = (const void*)(Jb + (size_t)(MoBlk + row) * NRB + ko + cu * 8);
      async_ld16(base + BBYTES + row * ASTRIDE + cu * 16, g);
    }
  };

  v8f acc[2][4];
#pragma unroll
  for (int a = 0; a < 2; ++a)
#pragma unroll
    for (int b = 0; b < 4; ++b)
#pragma unroll
      for (int e = 0; e < 8; ++e) acc[a][b][e] = 0.0f;

  const int mrow = lane & 15;
  const int half = lane >> 4;

  // Consume one staged K-chunk from LDS.
  auto compute = [&](int buf) {
    const unsigned char* Bs = &sm[buf][0];
    const unsigned char* As = &sm[buf][BBYTES];
    v16bf afr[2];
#pragma unroll
    for (int mi = 0; mi < 2; ++mi) {
      const unsigned char* ab = As + (wm * 32 + mi * 16 + mrow) * ASTRIDE + half * 16;
      v8bf lo = *(const v8bf*)(ab);          // K = half*8 + 0..7
      v8bf hi = *(const v8bf*)(ab + 32);     // K = 16 + half*8 + 0..7
      v16bf a;
#pragma unroll
      for (int e = 0; e < 8; ++e) { a[e] = lo[e]; a[8 + e] = hi[e]; }
      afr[mi] = a;
    }
    const unsigned char* bcol = Bs + (half * 8) * BSTRIDE + wn * 128 + mrow * 2;   // n = lane%16, K = half*8 + e / 16 + half*8 + e
#pragma unroll
    for (int j = 0; j < 4; ++j) {
      v16bf b;
#pragma unroll
      for (int e = 0; e < 8; ++e) {
        b[e]     = *(const __bf16*)(bcol + e * BSTRIDE + j * 32);
        b[8 + e] = *(const __bf16*)(bcol + (16 + e) * BSTRIDE + j * 32);
      }
#pragma unroll
      for (int mi = 0; mi < 2; ++mi)
        acc[mi][j] = __builtin_amdgcn_wmma_f32_16x16x32_bf16(
            false, afr[mi], false, b, (short)0, acc[mi][j], false, false);
    }
  };

  // Double-buffered pipeline: copy(k+1) overlaps WMMA(k).
  stage(0, 0);
  int cur = 0;
  for (int ko = 0; ko < NRB - 32; ko += 32) {
    stage(ko + 32, cur ^ 1);
    asm volatile("s_wait_asynccnt 5" ::: "memory");   // current buffer landed
    __syncthreads();
    compute(cur);
    __syncthreads();                                  // safe to overwrite cur next
    cur ^= 1;
  }
  asm volatile("s_wait_asynccnt 0" ::: "memory");
  __syncthreads();
  compute(cur);

  // Epilogue: p = J@x + h ; momentum update ; phi_s ; psi_s damping
  const int ncol0 = No + (lane & 15);
#pragma unroll
  for (int mi = 0; mi < 2; ++mi) {
#pragma unroll
    for (int v = 0; v < 8; ++v) {
      const int i = Mo + mi * 16 + half * 8 + v;
      const float hv = hvec[i];
#pragma unroll
      for (int j = 0; j < 4; ++j) {
        const size_t idx = (size_t)i * BSZ + ncol0 + j * 16;
        float p  = acc[mi][j][v] + hv;
        float xo = X[idx];
        float yv = YV[idx];
        yv = fmaf(dk, fmaf(-om_ak, xo, ec0 * p), yv);
        float xn = fmaf(dk, yv, xo);
        xn = phi_s(xn);
        yv *= (1.0f - psi_s(xn));
        X[idx] = xn;
        YV[idx] = yv;
        xbdst_u[idx] = f2bf(xn);
      }
    }
  }
}

// out[b*NRB + i] = X[i*BSZ + b]
__global__ __launch_bounds__(1024) void k_transpose(const float* __restrict__ X,
                                                    float* __restrict__ out) {
  __shared__ float tile[32][33];
  int i = blockIdx.y * 32 + threadIdx.y;   // row of X
  int b = blockIdx.x * 32 + threadIdx.x;   // col of X
  tile[threadIdx.y][threadIdx.x] = X[(size_t)i * BSZ + b];
  __syncthreads();
  int ob = blockIdx.x * 32 + threadIdx.y;
  int oi = blockIdx.y * 32 + threadIdx.x;
  out[(size_t)ob * NRB + oi] = tile[threadIdx.x][threadIdx.y];
}

// ---------------- Host launcher ----------------
extern "C" void kernel_launch(void* const* d_in, const int* in_sizes, int n_in,
                              void* d_out, int out_size, void* d_ws, size_t ws_size,
                              hipStream_t stream) {
  const float* Hr    = (const float*)d_in[0];
  const float* Hi    = (const float*)d_in[1];
  const float* yr    = (const float*)d_in[2];
  const float* yi    = (const float*)d_in[3];
  const float* Delta = (const float*)d_in[4];
  const float* eta   = (const float*)d_in[5];
  const float* lam   = (const float*)d_in[6];
  const float* x0    = (const float*)d_in[7];
  const float* y0    = (const float*)d_in[8];
  (void)in_sizes; (void)n_in; (void)out_size; (void)ws_size;

  char* ws = (char*)d_ws;
  size_t off = 0;
  auto alloc = [&](size_t bytes) -> void* {
    off = (off + 255) & ~(size_t)255;
    void* p = ws + off;
    off += bytes;
    return p;
  };

  float*          Ht    = (float*)alloc((size_t)NHT * NHT * 4);
  float*          ytl   = (float*)alloc((size_t)NHT * 4);
  float*          Aug   = (float*)alloc((size_t)NHT * AUGS * 4);
  float*          Kmat  = (float*)alloc((size_t)NHT * NHT * 4);
  float*          hvec  = (float*)alloc((size_t)NRB * 4);
  float*          stats = (float*)alloc(64);
  unsigned short* Jb    = (unsigned short*)alloc((size_t)NRB * NRB * 2);
  float*          X     = (float*)alloc((size_t)NRB * BSZ * 4);
  float*          YV    = (float*)alloc((size_t)NRB * BSZ * 4);
  unsigned short* XB0   = (unsigned short*)alloc((size_t)NRB * BSZ * 2);
  unsigned short* XB1   = (unsigned short*)alloc((size_t)NRB * BSZ * 2);

  // ---- Phase 1: LM setup ----
  k_build_ht<<<(NHT * NHT + 255) / 256, 256, 0, stream>>>(Hr, Hi, yr, yi, Ht, ytl);
  k_build_aug<<<(NHT * AUGS + 255) / 256, 256, 0, stream>>>(Ht, ytl, lam, Aug);
  k_gauss_jordan<<<1, 1024, 0, stream>>>(Aug);
  k_kmat<<<(NHT * NHT + 255) / 256, 256, 0, stream>>>(Ht, Aug, Kmat);
  k_stats_init<<<1, 1, 0, stream>>>(stats);
  k_expand_J<<<(NRB * NRB) / 256, 256, 0, stream>>>(Kmat, lam, Jb, stats);
  k_c0_h<<<1, 512, 0, stream>>>(Ht, Aug, lam, stats, hvec);
  k_init_xy<<<(int)(((size_t)NRB * BSZ + 255) / 256), 256, 0, stream>>>(x0, y0, X, YV, XB0);

  // ---- Phase 2: 50 fused DU-SB steps ----
  dim3 sgrid(BSZ / 256, NRB / 64);
  for (int t = 0; t < TSTEPS; ++t) {
    const float ak = (float)t / (float)(TSTEPS - 1);
    unsigned short* src = (t & 1) ? XB1 : XB0;
    unsigned short* dst = (t & 1) ? XB0 : XB1;
    k_sb_step<<<sgrid, 256, 0, stream>>>(Jb, src, dst, X, YV, hvec, stats,
                                         eta, Delta, t, ak);
  }

  // ---- Output: x.T (BSZ x NRB) ----
  dim3 tgrid(BSZ / 32, NRB / 32);
  k_transpose<<<tgrid, dim3(32, 32), 0, stream>>>(X, (float*)d_out);
}
